// VectorQuantizerEMA_55284819034586
// MI455X (gfx1250) — matrix operand order repaired and checked by reading, not compiled, not run on hardware
//
#include <hip/hip_runtime.h>
#include <hip/hip_bf16.h>

// Problem constants (from reference)
#define DIM      256
#define NEMB     1024
#define NROWS    65536          // 64*32*32
#define ROWS_PER_WAVE 32        // two 16-row A tiles resident per wave
#define ROWS_PER_BLOCK 256      // 8 waves x 32 rows
#define NTILES   (NEMB / 16)    // 64 column tiles
#define TILE_ELEMS 8192         // 16 emb x 256 dim x (hi+lo) bf16 = 16KB per tile
#define LOSS_OFF    16777216
#define IDX_OFF     16777217

typedef __attribute__((ext_vector_type(16))) __bf16 v16bf;
typedef __attribute__((ext_vector_type(8)))  float  v8f;

__device__ __forceinline__ __bf16 bf_trunc(float x) {
    unsigned u = __builtin_bit_cast(unsigned, x);
    unsigned short h = (unsigned short)(u >> 16);
    return __builtin_bit_cast(__bf16, h);
}
__device__ __forceinline__ float f_hi(float x) {
    unsigned u = __builtin_bit_cast(unsigned, x) & 0xffff0000u;
    return __builtin_bit_cast(float, u);
}
__device__ __forceinline__ void f4_to(float* t, float4 q) {
    t[0] = q.x; t[1] = q.y; t[2] = q.z; t[3] = q.w;
}
__device__ __forceinline__ void cvt16(const float* s, v16bf& vh, v16bf& vl) {
#pragma unroll
    for (int i = 0; i < 16; ++i) {
        float x = s[i];
        float h = f_hi(x);
        vh[i] = bf_trunc(x);
        vl[i] = bf_trunc(x - h);
    }
}

// One async 16B lane-copy global -> LDS (ASYNCcnt-tracked, gfx1250)
__device__ __forceinline__ void async_copy16(unsigned lds_off, const void* gptr) {
    asm volatile("global_load_async_to_lds_b128 %0, %1, off"
                 :: "v"(lds_off), "v"((unsigned long long)(uintptr_t)gptr)
                 : "memory");
}
__device__ __forceinline__ void wait_async0() {
    asm volatile("s_wait_asynccnt 0x0" ::: "memory");
}

// ---------------- Kernel 0a: codebook norms ----------------
__global__ void __launch_bounds__(256) vq_enorm_kernel(const float* __restrict__ E,
                                                       float* __restrict__ enorm) {
    int k = blockIdx.x * blockDim.x + threadIdx.x;
    if (k >= NEMB) return;
    const float4* p = reinterpret_cast<const float4*>(E + (size_t)k * DIM);
    float s = 0.f;
#pragma unroll 8
    for (int i = 0; i < DIM / 4; ++i) {
        float4 q = p[i];
        s += q.x * q.x + q.y * q.y + q.z * q.z + q.w * q.w;
    }
    enorm[k] = s;
}

// ---------------- Kernel 0b: pre-convert codebook to bf16 hi/lo in B layout ----
// wsB tile j (16KB): chunk c at j*8192 + c*1024; hi at +lane*16, lo at +512+lane*16.
__global__ void __launch_bounds__(256) vq_packB_kernel(const float* __restrict__ E,
                                                       unsigned short* __restrict__ wsB) {
    const int g    = blockIdx.x * 256 + threadIdx.x;   // 64*8*32 = 16384 threads
    const int j    = g >> 8;
    const int c    = (g >> 5) & 7;
    const int lane = g & 31;
    const int half = lane >> 4;
    const int ml   = lane & 15;
    const float* src = E + (size_t)(j * 16 + ml) * DIM + c * 32 + 8 * half;
    float t[16];
    const float4* p = reinterpret_cast<const float4*>(src);
    const float4* p1 = reinterpret_cast<const float4*>(src + 16);
    f4_to(t + 0, p[0]); f4_to(t + 4, p[1]); f4_to(t + 8, p1[0]); f4_to(t + 12, p1[1]);
    unsigned short* dhi = wsB + (size_t)j * TILE_ELEMS + c * 1024 + lane * 16;
    unsigned short* dlo = dhi + 512;
#pragma unroll
    for (int i = 0; i < 16; ++i) {
        unsigned u = __builtin_bit_cast(unsigned, t[i]);
        float hf   = __builtin_bit_cast(float, u & 0xffff0000u);
        dhi[i] = (unsigned short)(u >> 16);
        dlo[i] = (unsigned short)(__builtin_bit_cast(unsigned, t[i] - hf) >> 16);
    }
}

// ---------------- Kernel 1: WMMA distance GEMM + row argmin ----------------
// Wave owns 32 rows of X resident in regs (two 16-row bf16 hi/lo A tiles).
// Pre-converted B tiles are async-copied to LDS (double-buffered) per block and
// shared by all 8 waves: each 64B/lane B fetch now feeds 6 WMMAs (two
// independent accumulator chains). Lane-local argmin; one butterfly at the end.
__global__ void __launch_bounds__(256) vq_argmin_kernel(const float* __restrict__ X,
                                                        const unsigned short* __restrict__ wsB,
                                                        const float* __restrict__ enorm,
                                                        int* __restrict__ idx_i,
                                                        float* __restrict__ idx_f) {
    __shared__ __align__(16) unsigned short ldsB[2][TILE_ELEMS];   // 2 x 16KB
    const int tid  = threadIdx.x;
    const int wave = tid >> 5;
    const int lane = tid & 31;
    const int half = lane >> 4;
    const int ml   = lane & 15;
    const int rowBase = blockIdx.x * ROWS_PER_BLOCK + wave * ROWS_PER_WAVE;

    // ---- Preload + convert two resident A tiles (once; amortized over 64 tiles) ----
    v16bf a_hi0[8], a_lo0[8], a_hi1[8], a_lo1[8];
    {
        const float* xrow0 = X + (size_t)(rowBase + ml) * DIM;
        const float* xrow1 = X + (size_t)(rowBase + 16 + ml) * DIM;
#pragma unroll
        for (int c = 0; c < 8; ++c) {
            const int kb = 32 * c;
            float t[16];
            const float4* p0 = reinterpret_cast<const float4*>(xrow0 + kb + 8 * half);
            const float4* p1 = reinterpret_cast<const float4*>(xrow0 + kb + 16 + 8 * half);
            f4_to(t + 0, p0[0]); f4_to(t + 4, p0[1]); f4_to(t + 8, p1[0]); f4_to(t + 12, p1[1]);
            cvt16(t, a_hi0[c], a_lo0[c]);
            const float4* q0 = reinterpret_cast<const float4*>(xrow1 + kb + 8 * half);
            const float4* q1 = reinterpret_cast<const float4*>(xrow1 + kb + 16 + 8 * half);
            f4_to(t + 0, q0[0]); f4_to(t + 4, q0[1]); f4_to(t + 8, q1[0]); f4_to(t + 12, q1[1]);
            cvt16(t, a_hi1[c], a_lo1[c]);
        }
    }

    // Kick off async copy of tile 0 into buffer 0 (64B per thread = 4 x 16B)
    {
        const unsigned short* src = wsB + tid * 32;
        const unsigned dst = (unsigned)(uintptr_t)&ldsB[0][0] + tid * 64;
#pragma unroll
        for (int i = 0; i < 4; ++i) async_copy16(dst + 16 * i, src + 8 * i);
    }

    float best_v0[8], best_v1[8];
    int   best_i0[8], best_i1[8];
#pragma unroll
    for (int r = 0; r < 8; ++r) {
        best_v0[r] = 3.0e38f; best_i0[r] = 0;
        best_v1[r] = 3.0e38f; best_i1[r] = 0;
    }

#pragma unroll 1
    for (int j = 0; j < NTILES; ++j) {
        wait_async0();          // my slice of tile j has landed in LDS
        __syncthreads();        // everyone's slice visible
        if (j + 1 < NTILES) {   // prefetch next tile into the other buffer
            const unsigned short* src = wsB + (size_t)(j + 1) * TILE_ELEMS + tid * 32;
            const unsigned dst = (unsigned)(uintptr_t)&ldsB[(j + 1) & 1][0] + tid * 64;
#pragma unroll
            for (int i = 0; i < 4; ++i) async_copy16(dst + 16 * i, src + 8 * i);
        }

        const unsigned short* bbuf = &ldsB[j & 1][0];
        v8f acc0 = {}, acc1 = {};
#pragma unroll
        for (int c = 0; c < 8; ++c) {
            v16bf bh = *reinterpret_cast<const v16bf*>(bbuf + c * 1024 + lane * 16);
            v16bf bl = *reinterpret_cast<const v16bf*>(bbuf + c * 1024 + 512 + lane * 16);
            // Two independent accumulation chains share each B fetch.
            acc0 = __builtin_amdgcn_wmma_f32_16x16x32_bf16(false, a_hi0[c], false, bh,
                                                           (short)0, acc0, false, false);
            acc1 = __builtin_amdgcn_wmma_f32_16x16x32_bf16(false, a_hi1[c], false, bh,
                                                           (short)0, acc1, false, false);
            acc0 = __builtin_amdgcn_wmma_f32_16x16x32_bf16(false, a_hi0[c], false, bl,
                                                           (short)0, acc0, false, false);
            acc1 = __builtin_amdgcn_wmma_f32_16x16x32_bf16(false, a_hi1[c], false, bl,
                                                           (short)0, acc1, false, false);
            acc0 = __builtin_amdgcn_wmma_f32_16x16x32_bf16(false, a_lo0[c], false, bh,
                                                           (short)0, acc0, false, false);
            acc1 = __builtin_amdgcn_wmma_f32_16x16x32_bf16(false, a_lo1[c], false, bh,
                                                           (short)0, acc1, false, false);
        }
        // Lane-local running argmin: this lane always owns column j*16 + ml.
        const int   col = j * 16 + ml;
        const float en  = enorm[col];
#pragma unroll
        for (int r = 0; r < 8; ++r) {
            const float v0 = __builtin_fmaf(-2.0f, acc0[r], en);
            if (v0 < best_v0[r]) { best_v0[r] = v0; best_i0[r] = col; }
            const float v1 = __builtin_fmaf(-2.0f, acc1[r], en);
            if (v1 < best_v1[r]) { best_v1[r] = v1; best_i1[r] = col; }
        }
        __syncthreads();        // all waves done reading buf[j&1] before overwrite
    }

    // Single cross-lane butterfly per row (within each 16-lane half).
#pragma unroll
    for (int r = 0; r < 8; ++r) {
        float v = best_v0[r];
        int   i = best_i0[r];
#pragma unroll
        for (int s = 8; s >= 1; s >>= 1) {
            const float ov = __shfl_xor(v, s, 32);
            const int   oi = __shfl_xor(i, s, 32);
            if (ov < v || (ov == v && oi < i)) { v = ov; i = oi; }
        }
        best_i0[r] = i;
        float v1 = best_v1[r];
        int   i1 = best_i1[r];
#pragma unroll
        for (int s = 8; s >= 1; s >>= 1) {
            const float ov = __shfl_xor(v1, s, 32);
            const int   oi = __shfl_xor(i1, s, 32);
            if (ov < v1 || (ov == v1 && oi < i1)) { v1 = ov; i1 = oi; }
        }
        best_i1[r] = i1;
    }
    if (ml == 0) {
#pragma unroll
        for (int r = 0; r < 8; ++r) {
            const int row0 = rowBase + r + 8 * half;        // tile 0: rows 0..15
            idx_i[row0] = best_i0[r];
            idx_f[row0] = (float)best_i0[r];
            const int row1 = rowBase + 16 + r + 8 * half;   // tile 1: rows 16..31
            idx_i[row1] = best_i1[r];
            idx_f[row1] = (float)best_i1[r];
        }
    }
}

// ---------------- Kernel 2: gather quantized + per-block SSE partial ----------------
__global__ void __launch_bounds__(256) vq_gather_kernel(const float* __restrict__ X,
                                                        const float* __restrict__ E,
                                                        const int* __restrict__ idx_i,
                                                        float* __restrict__ quant,
                                                        float* __restrict__ partial) {
    __shared__ float red[256];
    const int t = threadIdx.x;
    const size_t base = (size_t)blockIdx.x * 4096 + (size_t)t * 16;
    const int row = (int)(base >> 8);
    const int d   = (int)(base & 255);
    const int k   = idx_i[row];
    const float4* xp = reinterpret_cast<const float4*>(X + base);
    const float4* ep = reinterpret_cast<const float4*>(E + (size_t)k * DIM + d);
    float4* op = reinterpret_cast<float4*>(quant + base);
    float s = 0.f;
#pragma unroll
    for (int i = 0; i < 4; ++i) {
        float4 q = ep[i];
        float4 x = xp[i];
        op[i] = q;
        float dx = q.x - x.x, dy = q.y - x.y, dz = q.z - x.z, dw = q.w - x.w;
        s += dx * dx + dy * dy + dz * dz + dw * dw;
    }
    red[t] = s;
    __syncthreads();
#pragma unroll
    for (int off = 128; off > 0; off >>= 1) {
        if (t < off) red[t] += red[t + off];
        __syncthreads();
    }
    if (t == 0) partial[blockIdx.x] = red[0];
}

// ---------------- Kernel 3: deterministic final loss reduction ----------------
__global__ void __launch_bounds__(256) vq_loss_kernel(const float* __restrict__ partial,
                                                      float* __restrict__ loss_out) {
    __shared__ float red[256];
    const int t = threadIdx.x;
    float s = 0.f;
    for (int i = t; i < 4096; i += 256) s += partial[i];
    red[t] = s;
    __syncthreads();
#pragma unroll
    for (int off = 128; off > 0; off >>= 1) {
        if (t < off) red[t] += red[t + off];
        __syncthreads();
    }
    if (t == 0) loss_out[0] = red[0] * (1.0f / ((float)NROWS * (float)DIM));
}

extern "C" void kernel_launch(void* const* d_in, const int* in_sizes, int n_in,
                              void* d_out, int out_size, void* d_ws, size_t ws_size,
                              hipStream_t stream) {
    (void)in_sizes; (void)n_in; (void)out_size; (void)ws_size;
    const float* X = (const float*)d_in[0];   // [65536, 256]
    const float* E = (const float*)d_in[1];   // [1024, 256]

    float* out   = (float*)d_out;
    float* quant = out;                 // [0, 16777216)
    float* loss  = out + LOSS_OFF;      // scalar
    float* idxf  = out + IDX_OFF;       // [65536] indices as float

    char* ws = (char*)d_ws;
    float* enorm   = (float*)ws;                                    // 1024 f32
    int*   idxi    = (int*)(ws + NEMB * sizeof(float));             // 65536 i32
    float* partial = (float*)(ws + (NEMB + NROWS) * sizeof(float)); // 4096 f32
    unsigned short* wsB =
        (unsigned short*)(ws + (NEMB + NROWS + 4096) * sizeof(float)); // 1MB bf16 hi/lo

    vq_enorm_kernel<<<NEMB / 256, 256, 0, stream>>>(E, enorm);
    vq_packB_kernel<<<NTILES * 8 * 32 / 256, 256, 0, stream>>>(E, wsB);
    vq_argmin_kernel<<<NROWS / ROWS_PER_BLOCK, 256, 0, stream>>>(X, wsB, enorm, idxi, idxf);
    vq_gather_kernel<<<4096, 256, 0, stream>>>(X, E, idxi, quant, partial);
    vq_loss_kernel<<<1, 256, 0, stream>>>(partial, loss);
}
